// WENONetwork_9474697855463
// MI455X (gfx1250) — matrix operand order repaired and checked by reading, not compiled, not run on hardware
//
#include <hip/hip_runtime.h>
#include <hip/hip_bf16.h>

typedef __attribute__((ext_vector_type(16))) _Float16 v16h;
typedef __attribute__((ext_vector_type(8)))  float    v8f;

#define THREADS 320          // 10 waves of 32
#define EXT 160              // extended positions computed per block
#define ROWS 164             // EXT + 2 guard rows on each side
#define OUT_PER_BLOCK 128

union Frag16 { v16h v; unsigned u[8]; };

// A-fragment (activations): M=16 positions x K=32 channels, f16.
// ISA layout (16-bit A 16x32): lane L holds M=L&15; kbase=(L>>4)*8;
// VGPR v<4: K=kbase+2v,+1 ; VGPR v>=4: K=16+kbase+2(v-4),+1
__device__ __forceinline__ v16h load_a_frag(const _Float16* act, int rowBase, int kpad, int k0, int lane) {
    const int m  = lane & 15;
    const int kb = (lane >> 4) * 8;
    const _Float16* row = act + (rowBase + m) * kpad + k0;
    Frag16 f;
#pragma unroll
    for (int v = 0; v < 8; ++v) {
        const int k = (v < 4) ? (kb + 2 * v) : (16 + kb + 2 * (v - 4));
        f.u[v] = *reinterpret_cast<const unsigned*>(row + k);
    }
    return f.v;
}

// B-fragment (weights): K=32 x N=16, f16, weight LDS layout [n][kpad].
// lane L holds N=L&15; kbase=(L>>4)*8; VGPR v<4: K=kbase+2v,+1 ; VGPR v>=4: K=16+kbase+2(v-4),+1
__device__ __forceinline__ v16h load_b_frag(const _Float16* wgt, int n0, int kpad, int k0, int lane) {
    const int n  = lane & 15;
    const int kb = (lane >> 4) * 8;
    const _Float16* row = wgt + (n0 + n) * kpad + k0;
    Frag16 f;
#pragma unroll
    for (int v = 0; v < 8; ++v)
        f.u[v] = *reinterpret_cast<const unsigned*>(row + ((v < 4) ? (kb + 2 * v) : (16 + kb + 2 * (v - 4))));
    return f.v;
}

// D tile: 32-bit C/D 16x16: VGPR r holds M=r (lanes 0-15) / M=8+r (lanes 16-31), N=lane&15.
// Apply bias + ELU, store f16 to dst[pos][chan].
__device__ __forceinline__ void store_d_elu(_Float16* dst, int rowBase, int kpad, int n0, v8f c, const float* bias, int lane) {
    const int n  = lane & 15;
    const int mh = (lane >> 4) * 8;
    const float b = bias[n0 + n];
#pragma unroll
    for (int r = 0; r < 8; ++r) {
        float x = c[r] + b;
        float y = (x > 0.0f) ? x : (__expf(x) - 1.0f);
        dst[(rowBase + r + mh) * kpad + n0 + n] = (_Float16)y;
    }
}

// Conv layer as WMMA GEMMs. KPS/KPD = src/dst padded channel counts,
// NT = N tiles (out-ch/16 padded), KST = K steps (in-ch/32 padded), TAPS = kernel width.
// Weight layout: [tap][NT*16][KST*32].
template<int KPS, int KPD, int NT, int KST, int TAPS>
__device__ __forceinline__ void conv_wmma(const _Float16* src, _Float16* dst,
                                          const _Float16* wgt, const float* bias,
                                          int posBase, int lane) {
    const int tapShift = (TAPS - 1) / 2;
#pragma unroll
    for (int nt = 0; nt < NT; ++nt) {
        const int n0 = nt * 16;
        v8f c = {};
#pragma unroll
        for (int t = 0; t < TAPS; ++t) {
            const int rowBase = posBase + 2 + t - tapShift;
            const _Float16* wt = wgt + (size_t)(t) * (NT * 16) * (KST * 32);
#pragma unroll
            for (int ks = 0; ks < KST; ++ks) {
                v16h a = load_a_frag(src, rowBase, KPS, ks * 32, lane);
                v16h b = load_b_frag(wt, n0, KST * 32, ks * 32, lane);
                c = __builtin_amdgcn_wmma_f32_16x16x32_f16(false, a, false, b, (short)0, c, false, false);
            }
        }
        store_d_elu(dst, posBase + 2, KPD, n0, c, bias, lane);
    }
}

__device__ __forceinline__ void zero_half_buf(_Float16* p, int nhalves, int tid) {
    unsigned* q = reinterpret_cast<unsigned*>(p);
    for (int i = tid; i < nhalves / 2; i += THREADS) q[i] = 0u;
}

__global__ __launch_bounds__(THREADS)
void weno_nn_kernel(const float* __restrict__ uu, const float* __restrict__ ep,
                    const float* __restrict__ w1, const float* __restrict__ b1,
                    const float* __restrict__ w2, const float* __restrict__ b2,
                    const float* __restrict__ w3, const float* __restrict__ b3,
                    const float* __restrict__ w4, const float* __restrict__ b4,
                    const float* __restrict__ w5, const float* __restrict__ b5,
                    const float* __restrict__ w6, const float* __restrict__ b6,
                    float* __restrict__ out, int n, int nout)
{
    // Activation ping-pong buffers (f16, [row][kpad]); buf1 max kpad 96, buf2 max kpad 64.
    __shared__ __align__(16) _Float16 buf1[ROWS * 96];
    __shared__ __align__(16) _Float16 buf2[ROWS * 64];
    // Padded f16 weights
    __shared__ __align__(16) _Float16 w2s[5 * 48 * 32];   // [tap][48][32]
    __shared__ __align__(16) _Float16 w3s[80 * 64];
    __shared__ __align__(16) _Float16 w4s[48 * 96];
    __shared__ __align__(16) _Float16 w5s[3 * 32 * 64];   // [tap][32][64]
    __shared__ float dif_s[ROWS];
    __shared__ float beta_s[EXT];
    __shared__ float w1s[20 * 5], w6s[20];
    __shared__ float b1s[20], b2s[48], b3s[80], b4s[48], b5s[32];
    __shared__ float b6s;

    const int tid  = threadIdx.x;
    const int lane = tid & 31;
    const int wave = tid >> 5;
    const long long jbase = (long long)blockIdx.x * OUT_PER_BLOCK;

    // ---- load + pad weights / biases into LDS (f16) ----
    for (int i = tid; i < 5 * 48 * 32; i += THREADS) {
        int t = i / (48 * 32), r = i % (48 * 32), nn = r / 32, k = r % 32;
        float v = (nn < 40 && k < 20) ? w2[nn * 100 + k * 5 + t] : 0.0f;
        w2s[i] = (_Float16)v;
    }
    for (int i = tid; i < 80 * 64; i += THREADS) {
        int nn = i / 64, k = i % 64;
        w3s[i] = (_Float16)((k < 40) ? w3[nn * 40 + k] : 0.0f);
    }
    for (int i = tid; i < 48 * 96; i += THREADS) {
        int nn = i / 96, k = i % 96;
        w4s[i] = (_Float16)((nn < 40 && k < 80) ? w4[nn * 80 + k] : 0.0f);
    }
    for (int i = tid; i < 3 * 32 * 64; i += THREADS) {
        int t = i / (32 * 64), r = i % (32 * 64), nn = r / 64, k = r % 64;
        float v = (nn < 20 && k < 40) ? w5[nn * 120 + k * 3 + t] : 0.0f;
        w5s[i] = (_Float16)v;
    }
    for (int i = tid; i < 100; i += THREADS) w1s[i] = w1[i];
    if (tid < 20) { w6s[tid] = w6[tid]; b1s[tid] = b1[tid]; }
    if (tid < 48) b2s[tid] = (tid < 40) ? b2[tid] : 0.0f;
    if (tid < 80) b3s[tid] = b3[tid];
    if (tid >= 80 && tid < 128) b4s[tid - 80] = (tid - 80 < 40) ? b4[tid - 80] : 0.0f;
    if (tid >= 128 && tid < 160) b5s[tid - 128] = (tid - 128 < 20) ? b5[tid - 128] : 0.0f;
    if (tid == 0) b6s = b6[0];

    // ---- avg_diff into LDS (ext position pe = di-2 maps to global p = jbase-16+pe) ----
    for (int di = tid; di < ROWS; di += THREADS) {
        long long p = jbase - 18 + di;
        float v = 0.0f;
        if (p >= 0 && p < n) {
            if (p == 0)            v = uu[1] - uu[0];
            else if (p == n - 1)   v = uu[n - 1] - uu[n - 2];
            else                   v = 0.5f * (uu[p + 1] - uu[p - 1]);
        }
        dif_s[di] = v;
    }
    zero_half_buf(buf1, ROWS * 96, tid);
    zero_half_buf(buf2, ROWS * 64, tid);
    __syncthreads();

    // ---- Layer 1 (1->20ch, k=5) in VALU; write buf1 [row][32] ----
    {
        int pe = tid >> 1, half = tid & 1;
        for (int c = 10 * half; c < 10 * half + 10; ++c) {
            float acc = b1s[c];
#pragma unroll
            for (int t = 0; t < 5; ++t) acc += w1s[c * 5 + t] * dif_s[pe + t];
            float y = (acc > 0.0f) ? acc : (__expf(acc) - 1.0f);
            buf1[(pe + 2) * 32 + c] = (_Float16)y;
        }
    }
    __syncthreads();

    const int posBase = wave * 16;   // each wave owns one M=16 strip of the 160 ext positions

    // ---- Layer 2: 20->40, k=5 : buf1(32) -> buf2(64) ----
    conv_wmma<32, 64, 3, 1, 5>(buf1, buf2, w2s, b2s, posBase, lane);
    __syncthreads();
    zero_half_buf(buf1, ROWS * 96, tid);
    __syncthreads();

    // ---- Layer 3: 40->80, k=1 : buf2(64) -> buf1(96) ----
    conv_wmma<64, 96, 5, 2, 1>(buf2, buf1, w3s, b3s, posBase, lane);
    __syncthreads();
    zero_half_buf(buf2, ROWS * 64, tid);
    __syncthreads();

    // ---- Layer 4: 80->40, k=1 : buf1(96) -> buf2(64) ----
    conv_wmma<96, 64, 3, 3, 1>(buf1, buf2, w4s, b4s, posBase, lane);
    __syncthreads();
    zero_half_buf(buf1, ROWS * 96, tid);
    __syncthreads();

    // ---- Layer 5: 40->20, k=3 : buf2(64) -> buf1(32) ----
    conv_wmma<64, 32, 2, 2, 3>(buf2, buf1, w5s, b5s, posBase, lane);
    __syncthreads();

    // ---- Layer 6: 20->1, k=1 + sigmoid + 0.1 ----
    if (tid < EXT) {
        float acc = b6s;
#pragma unroll
        for (int k = 0; k < 20; ++k) acc += w6s[k] * (float)buf1[(tid + 2) * 32 + k];
        beta_s[tid] = 1.0f / (1.0f + __expf(-acc)) + 0.1f;
    }
    __syncthreads();

    // ---- WENO5 flux ----
    if (tid < OUT_PER_BLOCK) {
        long long j = jbase + tid;
        if (j < nout) {
            const float eps = ep[0];
            float u1 = uu[j + 1], u2 = uu[j + 2], u3 = uu[j + 3];
            float u4 = uu[j + 4], u5 = uu[j + 5], u6 = uu[j + 6];
            float m0 = beta_s[tid + 18], m1 = beta_s[tid + 19], m2 = beta_s[tid + 20];
            const float c13 = 13.0f / 12.0f, c16 = 1.0f / 6.0f;
            float t;
            // plus branch (u shifted +1)
            float fp0 = (11.f * u4 - 7.f * u5 + 2.f * u6) * c16;
            float fp1 = (2.f * u3 + 5.f * u4 - u5) * c16;
            float fp2 = (-u2 + 5.f * u3 + 2.f * u4) * c16;
            t = u4 - 2.f * u5 + u6;        float bp0 = c13 * t * t;
            t = 3.f * u4 - 4.f * u5 + u6;  bp0 += 0.25f * t * t;
            t = u3 - 2.f * u4 + u5;        float bp1 = c13 * t * t;
            t = u3 - u5;                   bp1 += 0.25f * t * t;
            t = u2 - 2.f * u3 + u4;        float bp2 = c13 * t * t;
            t = u2 - 4.f * u3 + 3.f * u4;  bp2 += 0.25f * t * t;
            bp0 *= m0; bp1 *= m1; bp2 *= m2;
            // minus branch
            float fn0 = (11.f * u3 - 7.f * u4 + 2.f * u5) * c16;
            float fn1 = (2.f * u2 + 5.f * u3 - u4) * c16;
            float fn2 = (-u1 + 5.f * u2 + 2.f * u3) * c16;
            t = u3 - 2.f * u4 + u5;        float bn0 = c13 * t * t;
            t = 3.f * u3 - 4.f * u4 + u5;  bn0 += 0.25f * t * t;
            t = u2 - 2.f * u3 + u4;        float bn1 = c13 * t * t;
            t = u2 - u4;                   bn1 += 0.25f * t * t;
            t = u1 - 2.f * u2 + u3;        float bn2 = c13 * t * t;
            t = u1 - 4.f * u2 + 3.f * u3;  bn2 += 0.25f * t * t;
            bn0 *= m0; bn1 *= m1; bn2 *= m2;

            auto omega_flux = [&](float bb0, float bb1, float bb2, float f0, float f1, float f2) {
                float brs = bb2 - bb0; brs *= brs;
                float e0 = eps + bb0, e1 = eps + bb1, e2 = eps + bb2;
                float q0 = e0 * e0, q1 = e1 * e1, q2 = e2 * e2;
                float o0 = 0.1f / q0 * (brs + q0);
                float o1 = 0.6f / q1 * (brs + q1);
                float o2 = 0.3f / q2 * (brs + q2);
                float s = 1.0f / (o0 + o1 + o2);
                return (o0 * f0 + o1 * f1 + o2 * f2) * s;
            };
            float fluxp = omega_flux(bp0, bp1, bp2, fp0, fp1, fp2);
            float fluxn = omega_flux(bn0, bn1, bn2, fn0, fn1, fn2);
            out[j] = fluxp - fluxn;
        }
    }
}

extern "C" void kernel_launch(void* const* d_in, const int* in_sizes, int n_in,
                              void* d_out, int out_size, void* d_ws, size_t ws_size,
                              hipStream_t stream) {
    const float* uu = (const float*)d_in[0];
    const float* e  = (const float*)d_in[1];
    const float* w1 = (const float*)d_in[2];
    const float* b1 = (const float*)d_in[3];
    const float* w2 = (const float*)d_in[4];
    const float* b2 = (const float*)d_in[5];
    const float* w3 = (const float*)d_in[6];
    const float* b3 = (const float*)d_in[7];
    const float* w4 = (const float*)d_in[8];
    const float* b4 = (const float*)d_in[9];
    const float* w5 = (const float*)d_in[10];
    const float* b5 = (const float*)d_in[11];
    const float* w6 = (const float*)d_in[12];
    const float* b6 = (const float*)d_in[13];
    const int n = in_sizes[0];
    const int nout = out_size;
    const int blocks = (nout + OUT_PER_BLOCK - 1) / OUT_PER_BLOCK;
    weno_nn_kernel<<<blocks, THREADS, 0, stream>>>(uu, e, w1, b1, w2, b2, w3, b3,
                                                   w4, b4, w5, b5, w6, b6,
                                                   (float*)d_out, n, nout);
}
